// SpatialCirclePosition_40888088658044
// MI455X (gfx1250) — matrix operand order repaired and checked by reading, not compiled, not run on hardware
//
#include <hip/hip_runtime.h>

// Problem constants (match reference)
#define B_   16
#define N_   512
#define D_   768
#define H_   12
#define DK_  64
#define ND_  32
#define ROWS (B_ * N_)            // 8192 token rows
#define WELEM (D_ * D_)           // 589824 weight elems

typedef _Float16 h16;
typedef __attribute__((ext_vector_type(8)))  _Float16 v8h;
typedef __attribute__((ext_vector_type(16))) _Float16 v16h;
typedef __attribute__((ext_vector_type(8)))  float    v8f;

// ---------------------------------------------------------------------------
// Kernel 1: LayerNorm (eps=1e-5) -> f16 x
// ---------------------------------------------------------------------------
__global__ void __launch_bounds__(256)
k_layernorm(const float* __restrict__ feat, const float* __restrict__ gamma,
            const float* __restrict__ beta, h16* __restrict__ xh) {
    int row = blockIdx.x;                       // 0..ROWS-1
    const float* f = feat + (size_t)row * D_;
    __shared__ float red[256];
    float s = 0.f;
    for (int i = threadIdx.x; i < D_; i += 256) s += f[i];
    red[threadIdx.x] = s; __syncthreads();
    for (int o = 128; o > 0; o >>= 1) {
        if (threadIdx.x < o) red[threadIdx.x] += red[threadIdx.x + o];
        __syncthreads();
    }
    float mu = red[0] / (float)D_;
    __syncthreads();
    float v = 0.f;
    for (int i = threadIdx.x; i < D_; i += 256) { float d = f[i] - mu; v += d * d; }
    red[threadIdx.x] = v; __syncthreads();
    for (int o = 128; o > 0; o >>= 1) {
        if (threadIdx.x < o) red[threadIdx.x] += red[threadIdx.x + o];
        __syncthreads();
    }
    float inv = rsqrtf(red[0] / (float)D_ + 1e-5f);
    h16* out = xh + (size_t)row * D_;
    for (int i = threadIdx.x; i < D_; i += 256)
        out[i] = (h16)((f[i] - mu) * inv * gamma[i] + beta[i]);
}

// ---------------------------------------------------------------------------
// Kernel 2: f32 -> f16 convert (weights)
// ---------------------------------------------------------------------------
__global__ void __launch_bounds__(256)
k_cvt(const float* __restrict__ src, h16* __restrict__ dst, int n) {
    int i = blockIdx.x * 256 + threadIdx.x;
    if (i < n) dst[i] = (h16)src[i];
}

// ---------------------------------------------------------------------------
// Kernel 3: per-token 11x11 patch bin coordinates
// ---------------------------------------------------------------------------
__global__ void __launch_bounds__(256)
k_bins(const float* __restrict__ boxes, const int* __restrict__ isz,
       float* __restrict__ px, float* __restrict__ py) {
    int t = blockIdx.x * 256 + threadIdx.x;
    if (t >= ROWS) return;
    int b = t >> 9;
    float w = (float)isz[b * 4 + 0], h = (float)isz[b * 4 + 1];
    const float* bx = boxes + (size_t)t * 4;
    float x0 = bx[0] * w, y0 = bx[1] * h, x1 = bx[2] * w, y1 = bx[3] * h;
    float spw = floorf(w / 11.f), sph = floorf(h / 11.f);
    float cx = floorf((x0 + x1) * 0.5f), cy = floorf((y0 + y1) * 0.5f);
    int ix = 0, iy = 0; bool fx = false, fy = false;
    for (int j = 0; j < 11; j++) {
        float wl = j * spw, wh = (j + 1) * spw;
        float hl = j * sph, hh = (j + 1) * sph;
        if (!fx && wl <= cx && cx <= wh) { ix = j; fx = true; }
        if (!fy && hl <= cy && cy <= hh) { iy = j; fy = true; }
    }
    px[t] = (float)ix; py[t] = (float)iy;
}

// ---------------------------------------------------------------------------
// Kernel 4: WMMA GEMM  out[row,col] = sum_k X[row,k]*W[col,k] + bias[col]
//   One wave computes a 32x64 tile: per 32-wide K step, 2 A-frags + 4 B-frags
//   feed 8 v_wmma_f32_16x16x32_f16 (0.75 fragment loads per WMMA).
//   mode 0: Q  (f16 out, pre-scaled by 1/sqrt(DK))
//   mode 1: K  (f16 out)
//   mode 2: V  (f16 out, transposed per (b,h): Vt[(bh*DK+dk)*N + tok])
//   mode 3: O  (f32 out to d_out, non-temporal)
// ---------------------------------------------------------------------------
__global__ void __launch_bounds__(32)
k_gemm(const h16* __restrict__ X, const h16* __restrict__ Wh,
       const float* __restrict__ bias, int mode,
       h16* __restrict__ outh, float* __restrict__ outf) {
    int lane = threadIdx.x;
    int m = lane & 15, kh = lane >> 4;
    int col0 = blockIdx.x * 64, row0 = blockIdx.y * 32;

    const h16* arow0 = X + (size_t)(row0 + m) * D_;        // A rows, group 0
    const h16* arow1 = X + (size_t)(row0 + 16 + m) * D_;   // A rows, group 1
    const h16* brow0 = Wh + (size_t)(col0 + m) * D_;       // B cols j=0..3
    const h16* brow1 = Wh + (size_t)(col0 + 16 + m) * D_;
    const h16* brow2 = Wh + (size_t)(col0 + 32 + m) * D_;
    const h16* brow3 = Wh + (size_t)(col0 + 48 + m) * D_;

    v8f c00 = {}, c01 = {}, c02 = {}, c03 = {};
    v8f c10 = {}, c11 = {}, c12 = {}, c13 = {};

    for (int k0 = 0; k0 < D_; k0 += 32) {
        union { v16h v; v8h h2[2]; } a0, a1;
        // A 16x32 layout: lanes0-15 K=0..15, lanes16-31 K=16..31 (matches B)
        a0.h2[0] = *reinterpret_cast<const v8h*>(arow0 + k0 + kh * 16);
        a0.h2[1] = *reinterpret_cast<const v8h*>(arow0 + k0 + kh * 16 + 8);
        a1.h2[0] = *reinterpret_cast<const v8h*>(arow1 + k0 + kh * 16);
        a1.h2[1] = *reinterpret_cast<const v8h*>(arow1 + k0 + kh * 16 + 8);
        // B 32x16 layout: lanes0-15 K=0..15, lanes16-31 K=16..31 (col-major src)
        v16h b0 = *reinterpret_cast<const v16h*>(brow0 + k0 + kh * 16);
        v16h b1 = *reinterpret_cast<const v16h*>(brow1 + k0 + kh * 16);
        v16h b2 = *reinterpret_cast<const v16h*>(brow2 + k0 + kh * 16);
        v16h b3 = *reinterpret_cast<const v16h*>(brow3 + k0 + kh * 16);
        c00 = __builtin_amdgcn_wmma_f32_16x16x32_f16(false, a0.v, false, b0, (short)0, c00, false, false);
        c01 = __builtin_amdgcn_wmma_f32_16x16x32_f16(false, a0.v, false, b1, (short)0, c01, false, false);
        c02 = __builtin_amdgcn_wmma_f32_16x16x32_f16(false, a0.v, false, b2, (short)0, c02, false, false);
        c03 = __builtin_amdgcn_wmma_f32_16x16x32_f16(false, a0.v, false, b3, (short)0, c03, false, false);
        c10 = __builtin_amdgcn_wmma_f32_16x16x32_f16(false, a1.v, false, b0, (short)0, c10, false, false);
        c11 = __builtin_amdgcn_wmma_f32_16x16x32_f16(false, a1.v, false, b1, (short)0, c11, false, false);
        c12 = __builtin_amdgcn_wmma_f32_16x16x32_f16(false, a1.v, false, b2, (short)0, c12, false, false);
        c13 = __builtin_amdgcn_wmma_f32_16x16x32_f16(false, a1.v, false, b3, (short)0, c13, false, false);
    }

    float scale = (mode == 0) ? 0.125f : 1.0f;     // fold 1/sqrt(64) into Q
    v8f acc[2][4] = {{c00, c01, c02, c03}, {c10, c11, c12, c13}};
    for (int rg = 0; rg < 2; rg++) {
        for (int j = 0; j < 4; j++) {
            int col = col0 + j * 16 + m;
            float bb = bias[col];
            for (int r = 0; r < 8; r++) {
                int orow = row0 + rg * 16 + r + (kh ? 8 : 0);   // lanes16-31 -> M+8
                float val = (acc[rg][j][r] + bb) * scale;
                if (mode == 3) {
                    __builtin_nontemporal_store(val, outf + (size_t)orow * D_ + col);
                } else if (mode == 2) {
                    int bb_ = orow >> 9, tok = orow & 511;
                    int hh  = col >> 6,  dk  = col & 63;
                    outh[(((size_t)(bb_ * H_ + hh) * DK_ + dk) << 9) + tok] = (h16)val;
                } else {
                    outh[(size_t)orow * D_ + col] = (h16)val;
                }
            }
        }
    }
}

// ---------------------------------------------------------------------------
// Kernel 5: attention. One wave per (b, h, 16-query tile).
//   scores: 32 key tiles x 2 WMMA -> f32 LDS (stride 516, no bank conflicts)
//   bias:   dist_emb[int(2*sqrt(dx^2+dy^2))][h] from per-token patch bins
//   softmax: 2 lanes/row max+sum with shfl combine, per-row 1/sum in LDS,
//   then a COALESCED non-temporal writeback sweep (flat idx == global addr
//   offset) that also fills the f16 prob buffer ph[16][520] for phase 3.
//   att@V:  4 dk-tiles x 16 WMMA, f16 out per (b,tok,h*64+dk)
// ---------------------------------------------------------------------------
#define SCS 516
#define PHS 520
__global__ void __launch_bounds__(32)
k_attn(const h16* __restrict__ Qh, const h16* __restrict__ Kh,
       const h16* __restrict__ Vt, const float* __restrict__ px,
       const float* __restrict__ py, const float* __restrict__ demb,
       float* __restrict__ attOut, h16* __restrict__ Oh) {
    __shared__ float sc[16][SCS];
    __shared__ h16   ph[16][PHS];
    __shared__ float spx[N_], spy[N_], sde[ND_], sinv[16];
    int lane = threadIdx.x;
    int qt = blockIdx.x, h = blockIdx.y, b = blockIdx.z;

    for (int i = lane; i < N_; i += 32) { spx[i] = px[b * N_ + i]; spy[i] = py[b * N_ + i]; }
    sde[lane] = demb[lane * H_ + h];                 // dist_emb column h (ND==32)
    __syncthreads();

    int m = lane & 15, kh = lane >> 4;

    // ---- phase 1: S = (Q/sqrt(dk)) K^T  (16 x 512) ----
    int qrow = b * N_ + qt * 16 + m;
    const h16* qp = Qh + (size_t)qrow * D_ + h * DK_;
    union { v16h v; v8h h2[2]; } a0, a1;
    a0.h2[0] = *(const v8h*)(qp + kh * 16);       a0.h2[1] = *(const v8h*)(qp + kh * 16 + 8);
    a1.h2[0] = *(const v8h*)(qp + 32 + kh * 16);  a1.h2[1] = *(const v8h*)(qp + 40 + kh * 16);

    for (int kt = 0; kt < 32; kt++) {
        int krow = b * N_ + kt * 16 + m;                  // key owned by this lane
        const h16* kp = Kh + (size_t)krow * D_ + h * DK_;
        v16h b0 = *(const v16h*)(kp + kh * 16);
        v16h b1 = *(const v16h*)(kp + 32 + kh * 16);
        v8f c = {};
        c = __builtin_amdgcn_wmma_f32_16x16x32_f16(false, a0.v, false, b0, (short)0, c, false, false);
        c = __builtin_amdgcn_wmma_f32_16x16x32_f16(false, a1.v, false, b1, (short)0, c, false, false);
        for (int r = 0; r < 8; r++)
            sc[r + (kh ? 8 : 0)][kt * 16 + m] = c[r];
    }
    __syncthreads();

    // ---- phase 2a: bias + max + exp-sum (2 lanes per row) ----
    int qg = qt * 16 + m;
    float pqx = spx[qg], pqy = spy[qg];
    int c0 = kh * 256;
    float mx = -1e30f;
    for (int c = c0; c < c0 + 256; c++) {
        float dx = spx[c] - pqx, dy = spy[c] - pqy;
        int bin = (int)(sqrtf(dx * dx + dy * dy) * 2.0f);
        bin = bin < ND_ ? bin : ND_ - 1;
        float s = sc[m][c] + sde[bin];
        sc[m][c] = s;
        mx = fmaxf(mx, s);
    }
    mx = fmaxf(mx, __shfl_xor(mx, 16, 32));
    float sum = 0.f;
    for (int c = c0; c < c0 + 256; c++) {
        float e = __expf(sc[m][c] - mx);
        sc[m][c] = e;
        sum += e;
    }
    sum += __shfl_xor(sum, 16, 32);
    if (kh == 0) sinv[m] = 1.f / sum;
    __syncthreads();

    // ---- phase 2b: coalesced NT writeback of att + f16 probs to LDS ----
    // flat idx = row*512 + c maps linearly to global: abase + idx
    float* abase = attOut + (((size_t)(b * H_ + h) * N_ + qt * 16) * N_);
    for (int idx = lane; idx < 16 * N_; idx += 32) {
        int row = idx >> 9, c = idx & 511;
        float p = sc[row][c] * sinv[row];
        __builtin_nontemporal_store(p, abase + idx);
        ph[row][c] = (h16)p;
    }
    __syncthreads();

    // ---- phase 3: out = att @ V  (16 x 64) ----
    const h16* vbase = Vt + ((size_t)(b * H_ + h) * DK_) * (size_t)N_;
    for (int nt = 0; nt < 4; nt++) {
        int dk = nt * 16 + m;                     // dk column owned by this lane
        const h16* vp = vbase + (size_t)dk * N_;
        v8f c = {};
        for (int kc = 0; kc < 16; kc++) {
            union { v16h v; v8h h2[2]; } a;
            a.h2[0] = *(const v8h*)&ph[m][kc * 32 + kh * 16];
            a.h2[1] = *(const v8h*)&ph[m][kc * 32 + kh * 16 + 8];
            v16h bv = *(const v16h*)(vp + kc * 32 + kh * 16);
            c = __builtin_amdgcn_wmma_f32_16x16x32_f16(false, a.v, false, bv,
                                                       (short)0, c, false, false);
        }
        for (int r = 0; r < 8; r++) {
            int tok = qt * 16 + r + (kh ? 8 : 0);
            Oh[((size_t)(b * N_ + tok)) * D_ + h * DK_ + dk] = (h16)c[r];
        }
    }
}

// ---------------------------------------------------------------------------
// Host launcher
// ---------------------------------------------------------------------------
extern "C" void kernel_launch(void* const* d_in, const int* in_sizes, int n_in,
                              void* d_out, int out_size, void* d_ws, size_t ws_size,
                              hipStream_t stream) {
    const float* feat  = (const float*)d_in[0];
    const float* boxes = (const float*)d_in[1];
    const int*   isz   = (const int*)d_in[2];
    const float* Wq    = (const float*)d_in[3];
    const float* bq    = (const float*)d_in[4];
    const float* Wk    = (const float*)d_in[5];
    const float* bk    = (const float*)d_in[6];
    const float* Wv    = (const float*)d_in[7];
    const float* bv    = (const float*)d_in[8];
    const float* Wo    = (const float*)d_in[9];
    const float* bo    = (const float*)d_in[10];
    const float* gamma = (const float*)d_in[11];
    const float* beta  = (const float*)d_in[12];
    const float* demb  = (const float*)d_in[13];

    char* ws = (char*)d_ws;
    const size_t TOK16 = (size_t)ROWS * D_ * sizeof(h16);   // 12,582,912 B
    const size_t W16   = (size_t)WELEM * sizeof(h16);       //  1,179,648 B
    h16*   xh   = (h16*)(ws);
    h16*   qh   = (h16*)(ws + TOK16);
    h16*   kh_  = (h16*)(ws + 2 * TOK16);
    h16*   vt   = (h16*)(ws + 3 * TOK16);
    h16*   oh   = (h16*)(ws + 4 * TOK16);
    h16*   wq16 = (h16*)(ws + 5 * TOK16);
    h16*   wk16 = (h16*)(ws + 5 * TOK16 + W16);
    h16*   wv16 = (h16*)(ws + 5 * TOK16 + 2 * W16);
    h16*   wo16 = (h16*)(ws + 5 * TOK16 + 3 * W16);
    float* px   = (float*)(ws + 5 * TOK16 + 4 * W16);
    float* py   = px + ROWS;

    float* out = (float*)d_out;                               // (B,N,D)
    float* att = out + (size_t)B_ * N_ * D_;                  // (B,H,N,N)

    // prep
    k_layernorm<<<ROWS, 256, 0, stream>>>(feat, gamma, beta, xh);
    int cb = (WELEM + 255) / 256;
    k_cvt<<<cb, 256, 0, stream>>>(Wq, wq16, WELEM);
    k_cvt<<<cb, 256, 0, stream>>>(Wk, wk16, WELEM);
    k_cvt<<<cb, 256, 0, stream>>>(Wv, wv16, WELEM);
    k_cvt<<<cb, 256, 0, stream>>>(Wo, wo16, WELEM);
    k_bins<<<(ROWS + 255) / 256, 256, 0, stream>>>(boxes, isz, px, py);

    // projections (Q pre-scaled, V transposed per (b,h))
    dim3 gg(D_ / 64, ROWS / 32);
    k_gemm<<<gg, 32, 0, stream>>>(xh, wq16, bq, 0, qh, nullptr);
    k_gemm<<<gg, 32, 0, stream>>>(xh, wk16, bk, 1, kh_, nullptr);
    k_gemm<<<gg, 32, 0, stream>>>(xh, wv16, bv, 2, vt, nullptr);

    // attention
    k_attn<<<dim3(N_ / 16, H_, B_), 32, 0, stream>>>(qh, kh_, vt, px, py, demb, att, oh);

    // output projection
    k_gemm<<<gg, 32, 0, stream>>>(oh, wo16, bo, 3, nullptr, out);
}
